// prototypes_47656957116695
// MI455X (gfx1250) — matrix operand order repaired and checked by reading, not compiled, not run on hardware
//
#include <hip/hip_runtime.h>
#include <hip/hip_bf16.h>

// Problem constants (from reference): B=16, S=512, I=H=512, P=16, BS=8192
#define KDIM   512
#define PDIM   16
#define TILE_M 128         // rows per workgroup (8 waves, one 16-row M-block each)
#define PADK   520         // 512 + 8 halves padding -> 1040B row stride (4-bank lane skew)
#define NPASS  16          // 512 cols / (2 tiles * 16 cols) passes
#define BROWS  32          // W rows staged per pass (two 16-col output tiles)

typedef __attribute__((ext_vector_type(16))) _Float16     v16h;
typedef __attribute__((ext_vector_type(8)))  float        v8f;
typedef __attribute__((ext_vector_type(8)))  unsigned int v8u;

__device__ __forceinline__ unsigned int pk2h(float a, float b) {
  unsigned short ua = __builtin_bit_cast(unsigned short, (_Float16)a);
  unsigned short ub = __builtin_bit_cast(unsigned short, (_Float16)b);
  return (unsigned int)ua | ((unsigned int)ub << 16);
}

// A-fragment (16x32 f16): lane = row; VGPR0-7 hold K = k0 + half*16 .. +16
// (two aligned 16B LDS loads)
__device__ __forceinline__ v16h lds_frag_a(const _Float16* rowbase, int k0, int half) {
  uint4 lo = *(const uint4*)(rowbase + k0 + half * 16);
  uint4 hi = *(const uint4*)(rowbase + k0 + half * 16 + 8);
  v8u v;
  v[0] = lo.x; v[1] = lo.y; v[2] = lo.z; v[3] = lo.w;
  v[4] = hi.x; v[5] = hi.y; v[6] = hi.z; v[7] = hi.w;
  return __builtin_bit_cast(v16h, v);
}

// B-fragment (32x16 f16): lane = column; 16 contiguous K halves starting at
// koff (already includes +16 for upper half-wave) -> two aligned 16B LDS loads
__device__ __forceinline__ v16h lds_frag_b(const _Float16* colbase, int koff) {
  uint4 lo = *(const uint4*)(colbase + koff);
  uint4 hi = *(const uint4*)(colbase + koff + 8);
  v8u v;
  v[0] = lo.x; v[1] = lo.y; v[2] = lo.z; v[3] = lo.w;
  v[4] = hi.x; v[5] = hi.y; v[6] = hi.z; v[7] = hi.w;
  return __builtin_bit_cast(v16h, v);
}

// ---- register-double-buffered W staging -------------------------------------
// Each thread owns (row = tid>>3 of the 32-row tile, 64-float segment seg = tid&7).
// issue_w(): start the 16 global float4 loads for the *next* tile (no wait).
// store_w(): convert resident regs to f16 and write the LDS tile (waits loadcnt).
__device__ __forceinline__ void issue_w(const float* __restrict__ wrow, int seg,
                                        float4* R) {
#pragma unroll
  for (int j = 0; j < 8; ++j) {
    const int k = seg * 64 + j * 8;
    R[2 * j]     = *(const float4*)(wrow + k);
    R[2 * j + 1] = *(const float4*)(wrow + k + 4);
  }
}

__device__ __forceinline__ void store_w(const float4* R, _Float16* dstrow, int seg) {
#pragma unroll
  for (int j = 0; j < 8; ++j) {
    const int k = seg * 64 + j * 8;
    const float4 x = R[2 * j];
    const float4 y = R[2 * j + 1];
    uint4 u;
    u.x = pk2h(x.x, x.y); u.y = pk2h(x.z, x.w);
    u.z = pk2h(y.x, y.y); u.w = pk2h(y.z, y.w);
    *(uint4*)(dstrow + k) = u;
  }
}

__global__ __launch_bounds__(256) void proto_cos_kernel(
    const float* __restrict__ hidden, const float* __restrict__ W1,
    const float* __restrict__ W2, const float* __restrict__ protos,
    float* __restrict__ out) {
  __shared__ _Float16 Abuf[TILE_M * PADK];  // hidden tile, f16   (133,120 B)
  __shared__ _Float16 Hbuf[TILE_M * PADK];  // relu(h1) tile, f16 (133,120 B)
  __shared__ _Float16 Bbuf[BROWS * PADK];   // shared W tile      ( 33,280 B)

  const int tid    = threadIdx.x;
  const int wave   = tid >> 5;   // 0..7 -> M-block
  const int lane   = tid & 31;
  const int lane16 = lane & 15;
  const int half   = lane >> 4;
  const int p      = blockIdx.y;
  const int m0     = blockIdx.x * TILE_M;

  const float* W1p = W1 + (size_t)p * KDIM * KDIM;
  const float* W2p = W2 + (size_t)p * KDIM * KDIM;
  const float* pr  = protos + (size_t)p * KDIM;

  // staging role for W tiles
  const int wcol = tid >> 3;   // 0..31: row within the 32-row W tile
  const int wseg = tid & 7;    // 64-float segment within that row
  _Float16* bdst = Bbuf + wcol * PADK;
  float4 R[16];                // register double-buffer (64 VGPRs)

  // prefetch W1 tile 0 immediately (hides behind hidden staging + proto norm)
  issue_w(W1p + (size_t)wcol * KDIM, wseg, R);

  // ---- prototype norm (redundant per wave, reduced across full wave32) ----
  float ps = 0.f;
  for (int k = lane; k < KDIM; k += 32) { float v = pr[k]; ps += v * v; }
#pragma unroll
  for (int m = 16; m >= 1; m >>= 1) ps += __shfl_xor(ps, m, 32);
  const float pn = fmaxf(sqrtf(ps), 1e-6f);

  // ---- stage hidden tile (128 x 512 f32 -> f16 in LDS) ----
  {
    const int row = tid >> 1, hh = tid & 1;
    const float* src = hidden + (size_t)(m0 + row) * KDIM + hh * 256;
    _Float16* dst = Abuf + row * PADK + hh * 256;
#pragma unroll
    for (int j = 0; j < 32; ++j) {
      float4 x = *(const float4*)(src + j * 8);
      float4 y = *(const float4*)(src + j * 8 + 4);
      uint4 u;
      u.x = pk2h(x.x, x.y); u.y = pk2h(x.z, x.w);
      u.z = pk2h(y.x, y.y); u.w = pk2h(y.z, y.w);
      *(uint4*)(dst + j * 8) = u;
    }
  }

  const _Float16* arow  = Abuf + (wave * 16 + lane16) * PADK;
  const _Float16* hrow  = Hbuf + (wave * 16 + lane16) * PADK;
  const _Float16* bcol0 = Bbuf + lane16 * PADK;          // W rows  0..15 of tile
  const _Float16* bcol1 = Bbuf + (16 + lane16) * PADK;   // W rows 16..31 of tile

  // ---- layer 1: h1 = relu(hidden @ W1[p]^T), kept in LDS as f16 ----
  for (int nt = 0; nt < NPASS; ++nt) {
    const int n0 = nt * BROWS;
    __syncthreads();                       // previous consumers of Bbuf done
    store_w(R, bdst, wseg);                // resident regs -> LDS (waits loadcnt)
    __syncthreads();                       // Bbuf ready
    // prefetch next tile: W1 tile nt+1, or W2 tile 0 at the layer boundary
    const float* nxt = (nt + 1 < NPASS)
                           ? W1p + (size_t)((nt + 1) * BROWS + wcol) * KDIM
                           : W2p + (size_t)wcol * KDIM;
    issue_w(nxt, wseg, R);                 // overlaps with the WMMA loop below
    v8f c0 = {}, c1 = {};
#pragma unroll
    for (int ks = 0; ks < 16; ++ks) {
      v16h a  = lds_frag_a(arow, ks * 32, half);          // shared by both tiles
      v16h b0 = lds_frag_b(bcol0, ks * 32 + half * 16);
      v16h b1 = lds_frag_b(bcol1, ks * 32 + half * 16);
      c0 = __builtin_amdgcn_wmma_f32_16x16x32_f16(false, a, false, b0,
                                                  (short)0, c0, false, false);
      c1 = __builtin_amdgcn_wmma_f32_16x16x32_f16(false, a, false, b1,
                                                  (short)0, c1, false, false);
    }
    // D layout: VGPR r holds row r (lanes 0-15) / row r+8 (lanes 16-31), col = lane16
    _Float16* hdst = Hbuf + (wave * 16 + half * 8) * PADK + n0 + lane16;
#pragma unroll
    for (int r = 0; r < 8; ++r) {
      hdst[r * PADK]      = (_Float16)fmaxf(c0[r], 0.f);
      hdst[r * PADK + 16] = (_Float16)fmaxf(c1[r], 0.f);
    }
  }

  // ---- layer 2 + fused cosine reduction (h2 never stored) ----
  float dacc[8], sacc[8];
#pragma unroll
  for (int r = 0; r < 8; ++r) { dacc[r] = 0.f; sacc[r] = 0.f; }

  for (int nt = 0; nt < NPASS; ++nt) {
    const int n0 = nt * BROWS;
    __syncthreads();
    store_w(R, bdst, wseg);                // W2 tile nt (prefetched last pass)
    __syncthreads();
    if (nt + 1 < NPASS)                    // uniform branch, EXEC unchanged
      issue_w(W2p + (size_t)((nt + 1) * BROWS + wcol) * KDIM, wseg, R);
    v8f c0 = {}, c1 = {};
#pragma unroll
    for (int ks = 0; ks < 16; ++ks) {
      v16h a  = lds_frag_a(hrow, ks * 32, half);
      v16h b0 = lds_frag_b(bcol0, ks * 32 + half * 16);
      v16h b1 = lds_frag_b(bcol1, ks * 32 + half * 16);
      c0 = __builtin_amdgcn_wmma_f32_16x16x32_f16(false, a, false, b0,
                                                  (short)0, c0, false, false);
      c1 = __builtin_amdgcn_wmma_f32_16x16x32_f16(false, a, false, b1,
                                                  (short)0, c1, false, false);
    }
    const float w0 = pr[n0 + lane16];
    const float w1 = pr[n0 + 16 + lane16];
#pragma unroll
    for (int r = 0; r < 8; ++r) {
      float v0 = fmaxf(c0[r], 0.f);
      float v1 = fmaxf(c1[r], 0.f);
      dacc[r] += v0 * w0 + v1 * w1;   // dot(h2_row, proto)
      sacc[r] += v0 * v0 + v1 * v1;   // ||h2_row||^2
    }
  }

  // reduce the 16 columns held across each 16-lane half
#pragma unroll
  for (int r = 0; r < 8; ++r) {
#pragma unroll
    for (int m = 8; m >= 1; m >>= 1) {
      dacc[r] += __shfl_xor(dacc[r], m, 16);
      sacc[r] += __shfl_xor(sacc[r], m, 16);
    }
  }

  if (lane16 == 0) {   // lanes 0 and 16 each write 8 rows
#pragma unroll
    for (int r = 0; r < 8; ++r) {
      const int mrow = m0 + wave * 16 + half * 8 + r;
      const float xn = fmaxf(sqrtf(sacc[r]), 1e-6f);
      out[(size_t)mrow * PDIM + p] = dacc[r] / (xn * pn);
    }
  }
}

extern "C" void kernel_launch(void* const* d_in, const int* in_sizes, int n_in,
                              void* d_out, int out_size, void* d_ws, size_t ws_size,
                              hipStream_t stream) {
  const float* hidden = (const float*)d_in[0];  // [16,512,512]
  const float* W1     = (const float*)d_in[1];  // [16,512,512]
  const float* W2     = (const float*)d_in[2];  // [16,512,512]
  const float* protos = (const float*)d_in[3];  // [16,512]
  float* out = (float*)d_out;                   // [16,512,16]

  dim3 grid(8192 / TILE_M, PDIM);  // 64 M-tiles x 16 prototypes
  dim3 block(256);                 // 8 wave32
  proto_cos_kernel<<<grid, block, 0, stream>>>(hidden, W1, W2, protos, out);
}
